// blobTverskyLoss_2800318677067
// MI455X (gfx1250) — matrix operand order repaired and checked by reading, not compiled, not run on hardware
//
#include <hip/hip_runtime.h>
#include <hip/hip_bf16.h>

// blobTverskyLoss for MI455X (gfx1250, wave32).
// Single bandwidth-bound pass over x (B,2,D,H,W) + multi_label (B,D,H,W):
//  - labels stream memory -> LDS via GLOBAL_LOAD_ASYNC_TO_LDS_B128 (ASYNCcnt)
//  - v_wmma_f32_16x16x32_f16 acts as a 16-bin histogram engine:
//      A row0 = p0 (f16), row1 = 1;  B = one-hot(label)
//      => D row0 = per-label sum(p0), D row1 = per-label voxel count
//  - per-(batch,label) stats reduced via ds_add_f32 + global_atomic_add_f32
//  - tiny closed-form finalize kernel evaluates all 72 Tversky terms.

typedef __attribute__((ext_vector_type(16))) _Float16 v16h;
typedef __attribute__((ext_vector_type(8)))  _Float16 v8h;
typedef __attribute__((ext_vector_type(4)))  _Float16 v4h;
typedef __attribute__((ext_vector_type(8)))  float    v8f;
typedef __attribute__((ext_vector_type(4)))  float    v4f;

#define B_DIM 4
#define C_DIM 2
#define V_TOTAL (64 * 160 * 160)          // 1,638,400 voxels per sample
#define NLAB 16                           // padded label bins (real: 0..8)
#define WAVES_PER_BLOCK 8
#define BLOCK_SIZE 256
#define BLOCKS_PER_BATCH 80
#define WAVES_PER_BATCH (BLOCKS_PER_BATCH * WAVES_PER_BLOCK)  // 640
#define VOX_PER_ITER 128                  // 4 voxels per lane per iteration
#define ITERS (V_TOTAL / (WAVES_PER_BATCH * VOX_PER_ITER))    // 20 (exact)

#define TV_ALPHA 0.3f
#define TV_BETA  0.7f

// ---------------------------------------------------------------------------
// Pass 1: streaming histogram. grid = (80, B), block = 256 (8 wave32s).
// Every wave runs exactly ITERS uniform iterations of 128 voxels ->
// EXEC is all-ones at each WMMA (ISA 7.12 requirement), no tails.
// ---------------------------------------------------------------------------
__global__ __launch_bounds__(BLOCK_SIZE) void blob_hist_wmma_kernel(
    const float* __restrict__ x, const int* __restrict__ ml,
    float* __restrict__ ws)
{
  // Per-wave staging slabs (disjoint -> only wave-internal DS/ASYNC ordering)
  __shared__ __align__(16) _Float16 sh_p[WAVES_PER_BLOCK][VOX_PER_ITER];
  __shared__ __align__(16) int      sh_li[WAVES_PER_BLOCK][VOX_PER_ITER];
  __shared__ float sh_acc[NLAB * 2];

  const int tid  = threadIdx.x;
  const int lane = tid & 31;
  const int wv   = tid >> 5;
  const int b    = blockIdx.y;
  const int waveGlobal = blockIdx.x * WAVES_PER_BLOCK + wv;   // 0..639

  if (tid < NLAB * 2) sh_acc[tid] = 0.0f;
  __syncthreads();

  const float* xp0 = x + (size_t)b * C_DIM * V_TOTAL;   // class-0 logits
  const float* xp1 = xp0 + V_TOTAL;                     // class-1 logits
  const int*   mp  = ml + (size_t)b * V_TOTAL;

  const int n  = lane & 15;   // histogram bin (B column) / A row index
  const int hi = lane >> 4;   // which K-half this lane carries

  // LDS byte offset of this wave's label slab (shared aperture tag lives in
  // the upper 32 bits of the generic address; low 32 bits = LDS offset).
  const unsigned lds_lab = (unsigned)(uintptr_t)(&sh_li[wv][0]) + 16u * lane;

  const _Float16 r0 = (n == 0) ? (_Float16)1.0f : (_Float16)0.0f; // p0 row
  const _Float16 r1 = (n == 1) ? (_Float16)1.0f : (_Float16)0.0f; // ones row
  const _Float16 oneh  = (_Float16)1.0f;
  const _Float16 zeroh = (_Float16)0.0f;

  v8f acc = {};               // D: row0 = per-label sum(p0), row1 = counts

  for (int i = 0; i < ITERS; ++i) {
    const int vbase = (i * WAVES_PER_BATCH + waveGlobal) * VOX_PER_ITER;
    const int v4    = vbase + lane * 4;          // 4 consecutive voxels/lane

    // Drain previous iteration's DS reads before the async engine may
    // overwrite the label slab (ASYNC LDS writes are unordered vs DS).
    asm volatile("s_wait_dscnt 0" ::: "memory");

    // labels: memory -> LDS, 16B/lane (512B/wave, contiguous), ASYNCcnt.
    {
      const int goff = v4 * 4;                   // byte offset into mp
      asm volatile("global_load_async_to_lds_b128 %0, %1, %2"
                   :: "v"(lds_lab), "v"(goff), "s"(mp) : "memory");
    }

    // x: two coalesced b128 streams; sigmoid while the async load flies.
    const v4f a0 = *(const v4f*)&xp0[v4];
    const v4f a1 = *(const v4f*)&xp1[v4];
    v4h ph;
#pragma unroll
    for (int j = 0; j < 4; ++j) {
      // p0 = softmax(x)[0] = 1 / (1 + exp(x1 - x0))
      ph[j] = (_Float16)__builtin_amdgcn_rcpf(1.0f + __expf(a1[j] - a0[j]));
    }
    *(v4h*)&sh_p[wv][lane * 4] = ph;             // ds_store_b64

    asm volatile("s_wait_dscnt 0"    ::: "memory");  // p0 slab visible
    asm volatile("s_wait_asynccnt 0" ::: "memory");  // label slab visible

    // 4 sub-chunks of 32 voxels -> 4 WMMAs
#pragma unroll
    for (int c = 0; c < 4; ++c) {
      const int kb = c * 32;

      // ---- A (16x32 f16, contiguous k order, same as B): lane row M = lane&15;
      //      lanes 0-15 K = 0..15, lanes 16-31 K = 16..31.
      const v8h pa = *(const v8h*)&sh_p[wv][kb + hi * 16];
      const v8h pb = *(const v8h*)&sh_p[wv][kb + hi * 16 + 8];
      v16h A;
#pragma unroll
      for (int j = 0; j < 8; ++j) {
        A[j]     = pa[j] * r0 + r1;
        A[8 + j] = pb[j] * r0 + r1;
      }

      // ---- B (32x16 f16): lane column N = lane&15;
      //      lanes 0-15 hold K = 0..15, lanes 16-31 hold K = 16..31.
      //      B[k][n] = (label[k] == n), exact in f16.
      const int* lp = &sh_li[wv][kb + hi * 16];
      v16h Bt;
#pragma unroll
      for (int j = 0; j < 16; ++j) Bt[j] = (lp[j] == n) ? oneh : zeroh;

      // D += A x B
      acc = __builtin_amdgcn_wmma_f32_16x16x32_f16(
          /*neg_a=*/false, A, /*neg_b=*/false, Bt,
          /*c_mod=*/(short)0, acc, /*reuse_a=*/false, /*reuse_b=*/false);
    }
  }

  // D layout: VGPR r, lanes 0-15 -> M=r, N=lane. Rows >=2 of A are zero.
  if (lane < 16) {
    atomicAdd(&sh_acc[n * 2 + 0], acc[0]);   // sum p0 for label n
    atomicAdd(&sh_acc[n * 2 + 1], acc[1]);   // voxel count for label n
  }
  __syncthreads();
  if (tid < NLAB * 2) atomicAdd(&ws[b * (NLAB * 2) + tid], sh_acc[tid]);
}

// ---------------------------------------------------------------------------
// Zero the 128-float accumulator region each launch (harness doesn't re-poison)
// ---------------------------------------------------------------------------
__global__ void blob_zero_ws_kernel(float* __restrict__ ws) {
  const int i = threadIdx.x;
  if (i < B_DIM * NLAB * 2) ws[i] = 0.0f;
}

// ---------------------------------------------------------------------------
// Pass 2: closed-form finalize from per-(b,label) sufficient statistics.
// With p1 = 1-p0, S1[L] = N[L]-S0[L]; masking other blobs zeroes both logits
// there -> p' = 0.5 on those O voxels, p' = p elsewhere.
// ---------------------------------------------------------------------------
__global__ void blob_finalize_kernel(const float* __restrict__ ws,
                                     float* __restrict__ out) {
  if (threadIdx.x != 0 || blockIdx.x != 0) return;
  float main_acc = 0.0f;   // sum of main tversky over (b,c)
  float blob_acc = 0.0f;   // sum over (k,b) of CRIT_WEIGHT * (-(tv0+tv1)/2)
  for (int b = 0; b < B_DIM; ++b) {
    float S0[9], Nc[9];
    float sumS0 = 0.0f, sumN = 0.0f;
    for (int L = 0; L < 9; ++L) {
      S0[L] = ws[b * (NLAB * 2) + L * 2 + 0];
      Nc[L] = ws[b * (NLAB * 2) + L * 2 + 1];
      sumS0 += S0[L];
      sumN  += Nc[L];
    }
    const float S1_0 = Nc[0] - S0[0];          // sum p1 over background
    // main, class 0: g0 = (ml==0)
    {
      const float tp = S0[0];
      const float fp = sumS0 - tp;
      const float fn = Nc[0] - tp;
      main_acc += tp / fmaxf(tp + TV_ALPHA * fp + TV_BETA * fn, 1e-8f);
    }
    // main, class 1: g1 = (ml>0)
    {
      const float sumS1 = sumN - sumS0;
      const float tp = sumS1 - S1_0;
      const float fp = S1_0;
      const float fn = (sumN - Nc[0]) - tp;
      main_acc += tp / fmaxf(tp + TV_ALPHA * fp + TV_BETA * fn, 1e-8f);
    }
    const float posN = sumN - Nc[0];           // total blob voxels
    for (int k = 0; k < 8; ++k) {
      const int lab = k + 1;
      const float O   = posN - Nc[lab];        // other-blob voxels (p'=0.5)
      const float S1l = Nc[lab] - S0[lab];
      // class 1: g = (ml==lab)
      const float tp1 = S1l;
      const float fp1 = S1_0 + 0.5f * O;
      const float fn1 = Nc[lab] - tp1;
      const float tv1 = tp1 / fmaxf(tp1 + TV_ALPHA * fp1 + TV_BETA * fn1, 1e-8f);
      // class 0: g = (ml!=lab)
      const float tp0 = S0[0] + 0.5f * O;
      const float fp0 = S0[lab];
      const float fn0 = (sumN - Nc[lab]) - tp0;
      const float tv0 = tp0 / fmaxf(tp0 + TV_ALPHA * fp0 + TV_BETA * fn0, 1e-8f);
      blob_acc += -0.25f * (tv0 + tv1);        // 0.5 * (-(tv0+tv1)/2)
    }
  }
  const float main_loss = -(main_acc * (1.0f / 8.0f));   // -mean over (B,C)
  const float blob_loss = blob_acc * (1.0f / 32.0f);     // mean over (K,B)
  out[0] = 2.0f * (0.5f * main_loss) + 1.0f * blob_loss; // MAIN*CRIT + BLOB
}

// ---------------------------------------------------------------------------
extern "C" void kernel_launch(void* const* d_in, const int* in_sizes, int n_in,
                              void* d_out, int out_size, void* d_ws, size_t ws_size,
                              hipStream_t stream) {
  (void)in_sizes; (void)n_in; (void)out_size; (void)ws_size;
  const float* x  = (const float*)d_in[0];   // (4,2,64,160,160) f32
  const int*   ml = (const int*)d_in[1];     // (4,64,160,160) i32
  float* ws  = (float*)d_ws;                 // 128 floats used
  float* out = (float*)d_out;                // scalar loss

  blob_zero_ws_kernel<<<1, 128, 0, stream>>>(ws);
  dim3 grid(BLOCKS_PER_BATCH, B_DIM, 1);
  blob_hist_wmma_kernel<<<grid, BLOCK_SIZE, 0, stream>>>(x, ml, ws);
  blob_finalize_kernel<<<1, 32, 0, stream>>>(ws, out);
}
